// Attention_org_single_85074712199391
// MI455X (gfx1250) — matrix operand order repaired and checked by reading, not compiled, not run on hardware
//
#include <hip/hip_runtime.h>

// MI455X / gfx1250, wave32. Algebraically refactored attention:
//   S0[b,h,d,e] = sum_n Qc[b,n,h,d]*Kc[b,n,h,e]             (bf16 WMMA, big pass 1)
//   scores = (1/sqrt(960)) * Wq_h * S0 * Wk^T               (f32, tiny, L2)
//   probs  = softmax(instnorm(scores))                      (f32)
//   PW     = probs * Wv                                     (f32, tiny)
//   M^T[b,f,o] with f=h*240+e: M = Wo(:,c*4+h) * PW         (f32 -> bf16)
//   O1[b,n,o] = sum_f emb_all[b,n,f]*M^T[b,f,o]             (bf16 WMMA, big pass 2)
// ~42 GFLOP, ~390 MB HBM traffic -> needs matrix-core rates; bf16 WMMA w/ f32 acc.
// Workspace: ~51.2 MB.

typedef __attribute__((ext_vector_type(16))) __bf16 v16bf;
typedef __attribute__((ext_vector_type(8)))  __bf16 v8bf;
typedef __attribute__((ext_vector_type(8)))  float  v8f;

static __device__ __forceinline__ v16bf cat8(v8bf lo, v8bf hi) {
  return __builtin_shufflevector(lo, hi, 0,1,2,3,4,5,6,7,8,9,10,11,12,13,14,15);
}

// ---------------------------------------------------------------------------
// Kernel 1: S0 partials. grid (32 bh, 8 nsplit), 256 threads (8 waves).
// Each wave owns a 16-row (d) strip x 240 cols (e): 15 WMMA accum tiles.
// A fragment (16x32 bf16): lane m=l%16, g=l/16; halves k = 8g+0..7, 16+8g+0..7
//   -> contiguous in LDS Abuf[d][k] (transposed stage of emb1 tile).
// B fragment (32x16 bf16): lane n=l%16, g=l/16; k = 8g+0..7, 16+8g+0..7 -> strided column of Bbuf[k][e].
// ---------------------------------------------------------------------------
__global__ __launch_bounds__(256) void k_s0(const float* __restrict__ emb1,
                                            const float* __restrict__ emb_all,
                                            float* __restrict__ s0part)
{
  __shared__ __attribute__((aligned(16))) __bf16 Abuf[128][40];   // [d][k] + pad
  __shared__ __attribute__((aligned(16))) __bf16 Bbuf[32][248];   // [k][e] + pad

  const int bh = blockIdx.x, b = bh >> 2, h = bh & 3;
  const int ns = blockIdx.y;
  const int tid = threadIdx.x;
  const int lane = tid & 31, wv = tid >> 5;
  const int g = lane >> 4, mr = lane & 15;

  v8f acc[15];
#pragma unroll
  for (int t = 0; t < 15; ++t) acc[t] = 0.f;

  const size_t e1base = ((size_t)b * 4096 + (size_t)ns * 512) * 512 + (size_t)h * 128;
  const size_t eabase = ((size_t)b * 4096 + (size_t)ns * 512) * 960 + (size_t)h * 240;

  for (int kk = 0; kk < 512; kk += 32) {
    // stage emb1 tile [32 n x 128 d] -> Abuf[d][n] (transpose), f32 -> bf16
    for (int t0 = tid; t0 < 32 * 128; t0 += 256) {
      int i = t0 >> 7, j = t0 & 127;                 // i = n-local, j = d
      float v = emb1[e1base + (size_t)(kk + i) * 512 + j];
      Abuf[j][i] = (__bf16)v;
    }
    // stage emb_all tile [32 n x 240 e] -> Bbuf[n][e], f32 -> bf16
    for (int t0 = tid; t0 < 32 * 240; t0 += 256) {
      int k = t0 / 240, e = t0 - k * 240;
      float v = emb_all[eabase + (size_t)(kk + k) * 960 + e];
      Bbuf[k][e] = (__bf16)v;
    }
    if (kk + 32 < 512) {  // speculative prefetch of next tiles (global_prefetch_b8)
      __builtin_prefetch(emb1 + e1base + (size_t)(kk + 32 + (tid >> 3)) * 512 + (tid & 7) * 16, 0, 1);
      __builtin_prefetch(emb_all + eabase + (size_t)(kk + 32 + (tid >> 3)) * 960 + (tid & 7) * 30, 0, 1);
    }
    __syncthreads();

    const __bf16* ar = &Abuf[wv * 16 + mr][0];
    v16bf a = cat8(*(const v8bf*)(ar + 8 * g), *(const v8bf*)(ar + 16 + 8 * g));
#pragma unroll
    for (int t = 0; t < 15; ++t) {
      v16bf bt;
#pragma unroll
      for (int s = 0; s < 16; ++s) bt[s] = Bbuf[(s < 8 ? 8 * g + s : 16 + 8 * g + (s - 8))][16 * t + mr];
      acc[t] = __builtin_amdgcn_wmma_f32_16x16x32_bf16(false, a, false, bt,
                                                       (short)0, acc[t], false, false);
    }
    __syncthreads();
  }

  // C/D layout: lane l: col = l%16; VGPR v: row = v + 8*(l/16)
  float* out = s0part + ((size_t)bh * 8 + ns) * 30720;
#pragma unroll
  for (int t = 0; t < 15; ++t)
#pragma unroll
    for (int v = 0; v < 8; ++v)
      out[(wv * 16 + v + 8 * g) * 240 + 16 * t + mr] = acc[t][v];
}

// ---------------------------------------------------------------------------
// Kernel 2: per-(b,h) score pipeline, pure f32 (numerics-sensitive path).
// One 1024-thread block per (b,h); small matrices live in L2 via ws.
// ---------------------------------------------------------------------------
__global__ __launch_bounds__(1024) void k_probs(float* __restrict__ s0part,
                                                const float* __restrict__ Wq,
                                                const float* __restrict__ Wk,
                                                const float* __restrict__ Wv,
                                                float* __restrict__ tmpA,
                                                float* __restrict__ tmpB,
                                                float* __restrict__ pw)
{
  __shared__ float redS[32], redS2[32], stats[2];
  const int bh = blockIdx.x, h = bh & 3;
  const int tid = threadIdx.x;

  float* s0 = s0part + (size_t)bh * 8 * 30720;
  // phase 0: reduce the 8 N-split partials into split 0
  for (int idx = tid; idx < 30720; idx += 1024) {
    float s = s0[idx];
#pragma unroll
    for (int sp = 1; sp < 8; ++sp) s += s0[(size_t)sp * 30720 + idx];
    s0[idx] = s;
  }
  __threadfence(); __syncthreads();

  // phase 1: tmpA[c][e] = sum_d Wq[h][c][d] * S0[d][e]
  float* ta = tmpA + (size_t)bh * 30720;
  for (int r = 0; r < 30; ++r) {
    int idx = tid + (r << 10);
    int c = idx / 240, e = idx - c * 240;
    const float* wq = Wq + ((size_t)h * 128 + c) * 128;
    float s = 0.f;
    for (int d = 0; d < 128; ++d) s += wq[d] * s0[d * 240 + e];
    ta[idx] = s;
  }
  __threadfence(); __syncthreads();

  // phase 2: scores[c][k] = (1/sqrt(960)) * sum_e tmpA[c][e] * Wk[k][e]
  float* sc = tmpB + (size_t)bh * 30720;
  const float scale = 0.032274861218395144f;  // 1/sqrt(960)
  for (int r = 0; r < 30; ++r) {
    int idx = tid + (r << 10);
    int c = idx / 240, k = idx - c * 240;
    const float* tr = ta + c * 240;
    const float* wk = Wk + (size_t)k * 240;
    float s = 0.f;
    for (int e = 0; e < 240; ++e) s += tr[e] * wk[e];
    sc[idx] = s * scale;
  }
  __threadfence(); __syncthreads();

  // phase 3: InstanceNorm stats over all 30720 elems (biased var, eps=1e-5)
  float s = 0.f, s2 = 0.f;
  for (int r = 0; r < 30; ++r) { float v = sc[tid + (r << 10)]; s += v; s2 += v * v; }
#pragma unroll
  for (int off = 16; off > 0; off >>= 1) {
    s  += __shfl_down(s,  off, 32);
    s2 += __shfl_down(s2, off, 32);
  }
  const int lane = tid & 31, wv = tid >> 5;
  if (lane == 0) { redS[wv] = s; redS2[wv] = s2; }
  __syncthreads();
  if (tid < 32) {
    s = redS[tid]; s2 = redS2[tid];
#pragma unroll
    for (int off = 16; off > 0; off >>= 1) {
      s  += __shfl_down(s,  off, 32);
      s2 += __shfl_down(s2, off, 32);
    }
    if (tid == 0) {
      float mu = s * (1.f / 30720.f);
      float var = s2 * (1.f / 30720.f) - mu * mu;
      stats[0] = mu;
      stats[1] = rsqrtf(var + 1e-5f);
    }
  }
  __syncthreads();
  const float mu = stats[0], rin = stats[1];

  // phase 4: softmax over k per row c (128 rows), normalize inline
  if (tid < 128) {
    float* row = sc + tid * 240;
    float mx = -3.4e38f;
    for (int k = 0; k < 240; ++k) { float x = (row[k] - mu) * rin; mx = fmaxf(mx, x); }
    float den = 0.f;
    for (int k = 0; k < 240; ++k) den += __expf((row[k] - mu) * rin - mx);
    float inv = 1.f / den;
    for (int k = 0; k < 240; ++k) row[k] = __expf((row[k] - mu) * rin - mx) * inv;
  }
  __threadfence(); __syncthreads();

  // phase 5: PW[c][e] = sum_k probs[c][k] * Wv[k][e]
  float* pwr = pw + (size_t)bh * 30720;
  for (int r = 0; r < 30; ++r) {
    int idx = tid + (r << 10);
    int c = idx / 240, e = idx - c * 240;
    const float* pr = sc + c * 240;
    float sacc = 0.f;
    for (int k = 0; k < 240; ++k) sacc += pr[k] * Wv[(size_t)k * 240 + e];
    pwr[idx] = sacc;
  }
}

// ---------------------------------------------------------------------------
// Kernel 3: fold Wo: mT[b][h*240+e][o] = (bf16) sum_c Wo[o][c*4+h]*PW[bh][c][e]
// grid (32 bh, 240 e), 256 threads.
// ---------------------------------------------------------------------------
__global__ __launch_bounds__(256) void k_mmat(const float* __restrict__ pw,
                                              const float* __restrict__ Wo,
                                              __bf16* __restrict__ mT)
{
  __shared__ float col[128];
  const int bh = blockIdx.x, b = bh >> 2, h = bh & 3;
  const int e = blockIdx.y;
  const int tid = threadIdx.x;
  if (tid < 128) col[tid] = pw[(size_t)bh * 30720 + tid * 240 + e];
  __syncthreads();
#pragma unroll
  for (int rep = 0; rep < 2; ++rep) {
    int o = tid + rep * 256;
    const float* wr = Wo + (size_t)o * 512 + h;
    float s = 0.f;
    for (int c = 0; c < 128; ++c) s += col[c] * wr[c * 4];
    mT[((size_t)b * 960 + (size_t)h * 240 + e) * 512 + o] = (__bf16)s;
  }
}

// ---------------------------------------------------------------------------
// Kernel 4: O1[b][n][o] = sum_f emb_all[b][n][f] * mT[b][f][o], bf16 WMMA.
// grid (32 ntile, 4 otile, 8 b), 256 threads; 128x128 output tile per block.
// ---------------------------------------------------------------------------
__global__ __launch_bounds__(256) void k_out(const float* __restrict__ emb_all,
                                             const __bf16* __restrict__ mT,
                                             float* __restrict__ out)
{
  __shared__ __attribute__((aligned(16))) __bf16 Abuf[128][40];   // [n][f] + pad
  __shared__ __attribute__((aligned(16))) __bf16 Bbuf[32][136];   // [f][o] + pad

  const int nt = blockIdx.x, ot = blockIdx.y, b = blockIdx.z;
  const int tid = threadIdx.x;
  const int lane = tid & 31, wv = tid >> 5;
  const int g = lane >> 4, mr = lane & 15;

  v8f acc[8];
#pragma unroll
  for (int t = 0; t < 8; ++t) acc[t] = 0.f;

  const size_t abase = ((size_t)b * 4096 + (size_t)nt * 128) * 960;
  const size_t bbase = (size_t)b * 960 * 512 + (size_t)ot * 128;

  for (int f0 = 0; f0 < 960; f0 += 32) {
    for (int t0 = tid; t0 < 128 * 32; t0 += 256) {
      int i = t0 >> 5, j = t0 & 31;                 // i = n-local, j = f-local
      Abuf[i][j] = (__bf16)emb_all[abase + (size_t)i * 960 + f0 + j];
    }
    for (int t0 = tid; t0 < 32 * 128; t0 += 256) {
      int k = t0 >> 7, o = t0 & 127;
      Bbuf[k][o] = mT[bbase + (size_t)(f0 + k) * 512 + o];
    }
    if (f0 + 32 < 960) {
      __builtin_prefetch(emb_all + abase + (size_t)(tid >> 1) * 960 + f0 + 32 + (tid & 1) * 16, 0, 1);
    }
    __syncthreads();

    const __bf16* ar = &Abuf[wv * 16 + mr][0];
    v16bf a = cat8(*(const v8bf*)(ar + 8 * g), *(const v8bf*)(ar + 16 + 8 * g));
#pragma unroll
    for (int t = 0; t < 8; ++t) {
      v16bf bt;
#pragma unroll
      for (int s = 0; s < 16; ++s) bt[s] = Bbuf[(s < 8 ? 8 * g + s : 16 + 8 * g + (s - 8))][16 * t + mr];
      acc[t] = __builtin_amdgcn_wmma_f32_16x16x32_bf16(false, a, false, bt,
                                                       (short)0, acc[t], false, false);
    }
    __syncthreads();
  }

#pragma unroll
  for (int t = 0; t < 8; ++t)
#pragma unroll
    for (int v = 0; v < 8; ++v)
      out[((size_t)b * 4096 + nt * 128 + wv * 16 + v + 8 * g) * 512
          + ot * 128 + 16 * t + mr] = acc[t][v];
}

// ---------------------------------------------------------------------------
extern "C" void kernel_launch(void* const* d_in, const int* in_sizes, int n_in,
                              void* d_out, int out_size, void* d_ws, size_t ws_size,
                              hipStream_t stream)
{
  const float* emb1    = (const float*)d_in[0];
  const float* emb_all = (const float*)d_in[1];
  const float* Wq      = (const float*)d_in[2];
  const float* Wk      = (const float*)d_in[3];
  const float* Wv      = (const float*)d_in[4];
  const float* Wo      = (const float*)d_in[5];
  float* out = (float*)d_out;

  char* w = (char*)d_ws;
  float*  s0part = (float*)(w);               // 32*8*30720*4  = 31,457,280 B
  float*  tmpA   = (float*)(w + 31457280);    // 32*30720*4    =  3,932,160 B
  float*  tmpB   = (float*)(w + 35389440);    // 32*30720*4
  float*  pw     = (float*)(w + 39321600);    // 32*30720*4
  __bf16* mT     = (__bf16*)(w + 43253760);   // 8*960*512*2   =  7,864,320 B
  // total ws use: 51,118,080 B

  k_s0   <<<dim3(32, 8),    256,  0, stream>>>(emb1, emb_all, s0part);
  k_probs<<<dim3(32),       1024, 0, stream>>>(s0part, Wq, Wk, Wv, tmpA, tmpB, pw);
  k_mmat <<<dim3(32, 240),  256,  0, stream>>>(pw, Wo, mT);
  k_out  <<<dim3(32, 4, 8), 256,  0, stream>>>(emb_all, mT, out);
}
